// LossNet_42494406426743
// MI455X (gfx1250) — matrix operand order repaired and checked by reading, not compiled, not run on hardware
//
#include <hip/hip_runtime.h>

// ---------------------------------------------------------------------------
// Contrastive loss on MI455X (gfx1250, wave32, WMMA).
//
// x: (12288, 128) fp32.  bsz = 4096.  temp = 0.1.
// Pipeline:
//   1) normalize rows of x, convert to f16   -> ws.xh  [12288 x 128] f16
//   2) 7 fused GEMM+exp+rowsum passes via v_wmma_f32_16x16x32_f16
//      (branch-free epilogue, raw v_exp_f32; partials -> ws.P [7][4][4096])
//   3) diagonals via trivial per-row dot kernel -> ws.D [7][4096]
//   4) single-block scalar reduction -> d_out[0]
// ---------------------------------------------------------------------------

typedef _Float16 v16h __attribute__((ext_vector_type(16)));
typedef _Float16 v8h  __attribute__((ext_vector_type(8)));
typedef float    v8f  __attribute__((ext_vector_type(8)));

#define BSZ    4096
#define KDIM   128
#define NMAT   7
#define NCHUNK 4          // 4096 columns split into 4 chunks of 1024
#define LOG2E_OVER_TEMP 14.426950408889634f   // (1/0.1) * log2(e)

__device__ __constant__ int c_asel[NMAT] = {0, 1, 0, 0, 1, 2, 2}; // xx yy xy ax ay zx zy
__device__ __constant__ int c_bsel[NMAT] = {0, 1, 1, 2, 2, 0, 1};

// Raw v_exp_f32: argument is bounded in [-14.5, 14.5], no denorm guard needed.
__device__ __forceinline__ float fast_exp_sim(float sim) {
    return __builtin_amdgcn_exp2f(sim * LOG2E_OVER_TEMP);
}

// ---------------------------------------------------------------------------
// 1) Row normalization + f16 conversion. One 128-thread block per row.
// ---------------------------------------------------------------------------
__global__ __launch_bounds__(128) void normalize_f16(const float* __restrict__ x,
                                                     _Float16* __restrict__ xh) {
    const int row = blockIdx.x;
    const int t   = threadIdx.x;
    const float v = x[(size_t)row * KDIM + t];
    float ss = v * v;
    #pragma unroll
    for (int m = 16; m >= 1; m >>= 1) ss += __shfl_xor(ss, m, 32);
    __shared__ float red[4];
    const int wid = t >> 5, lane = t & 31;
    if (lane == 0) red[wid] = ss;
    __syncthreads();
    const float tot = red[0] + red[1] + red[2] + red[3];
    const float inv = 1.0f / fmaxf(sqrtf(tot), 1e-12f);
    xh[(size_t)row * KDIM + t] = (_Float16)(v * inv);
}

// ---------------------------------------------------------------------------
// 2) Fused GEMM + exp + rowsum (branch-free hot loop).
//    blockIdx.y = matmul id m (0..6)
//    blockIdx.x = rb*4 + chunk   (rb: 32 row-blocks of 128 rows; chunk: 0..3)
//    256 threads = 8 waves; wave w owns rows [rb*128 + w*16, +16),
//    columns [chunk*1024, +1024) as 64 tiles of 16.
// ---------------------------------------------------------------------------
__global__ __launch_bounds__(256) void gemm_expsum(const _Float16* __restrict__ xh,
                                                   float* __restrict__ P) {
    const int m     = blockIdx.y;
    const int rb    = blockIdx.x >> 2;
    const int chunk = blockIdx.x & 3;
    const int wave  = threadIdx.x >> 5;
    const int lane  = threadIdx.x & 31;
    const int h     = lane >> 4;   // half-wave selector
    const int n     = lane & 15;   // A: row-in-tile; B: col-in-tile

    const int row0     = rb * 128 + wave * 16;
    const int col0base = chunk * 1024;

    const _Float16* A = xh + (size_t)c_asel[m] * BSZ * KDIM;
    const _Float16* B = xh + (size_t)c_bsel[m] * BSZ * KDIM;

    // Load the wave's 16x128 A strip in WMMA A-operand layout.
    // 16-bit A 16x32: lane row = lane&15; halfword i -> K = kt*32+16h+i
    // (contiguous, matching B): two contiguous v8h chunks.
    v16h a[4];
    const _Float16* arow = A + (size_t)(row0 + n) * KDIM;
    #pragma unroll
    for (int kt = 0; kt < 4; ++kt) {
        const v8h lo = *(const v8h*)(arow + kt * 32 + 16 * h);
        const v8h hi = *(const v8h*)(arow + kt * 32 + 16 * h + 8);
        v16h av;
        #pragma unroll
        for (int i = 0; i < 8; ++i) { av[i] = lo[i]; av[i + 8] = hi[i]; }
        a[kt] = av;
    }

    float acc[8];
    #pragma unroll
    for (int v = 0; v < 8; ++v) acc[v] = 0.0f;

    for (int ct = 0; ct < 64; ++ct) {
        // B-operand 16-bit 32x16: lane col = lane&15; halfword i -> K = kt*32+16h+i
        // => one contiguous 32B v16h per K-tile. Distinct regs so loads clause.
        const _Float16* brow = B + (size_t)(col0base + ct * 16 + n) * KDIM + 16 * h;
        const v16h bv0 = *(const v16h*)(brow);
        const v16h bv1 = *(const v16h*)(brow + 32);
        const v16h bv2 = *(const v16h*)(brow + 64);
        const v16h bv3 = *(const v16h*)(brow + 96);

        v8f c = {};
        c = __builtin_amdgcn_wmma_f32_16x16x32_f16(false, a[0], false, bv0, (short)0, c, false, false);
        c = __builtin_amdgcn_wmma_f32_16x16x32_f16(false, a[1], false, bv1, (short)0, c, false, false);
        c = __builtin_amdgcn_wmma_f32_16x16x32_f16(false, a[2], false, bv2, (short)0, c, false, false);
        c = __builtin_amdgcn_wmma_f32_16x16x32_f16(false, a[3], false, bv3, (short)0, c, false, false);

        // Branch-free epilogue: exp(sim/temp) via raw v_exp_f32, accumulate.
        #pragma unroll
        for (int v = 0; v < 8; ++v)
            acc[v] += fast_exp_sim(c[v]);
    }

    // Reduce each rowsum across the 16 lanes sharing h (bit 4 untouched).
    #pragma unroll
    for (int v = 0; v < 8; ++v) {
        float s = acc[v];
        s += __shfl_xor(s, 1, 32);
        s += __shfl_xor(s, 2, 32);
        s += __shfl_xor(s, 4, 32);
        s += __shfl_xor(s, 8, 32);
        acc[v] = s;
    }
    if (n == 0) {
        // C/D layout: VGPR v of lane -> row M = v + 8h. Exclusive ownership.
        float* Pp = P + ((size_t)m * NCHUNK + chunk) * BSZ + row0 + 8 * h;
        #pragma unroll
        for (int v = 0; v < 8; ++v) Pp[v] = acc[v];
    }
}

// ---------------------------------------------------------------------------
// 3) Diagonals: d[m][i] = exp(dot(A_i, B_i)/temp). 7*4096 length-128 dots.
// ---------------------------------------------------------------------------
__global__ __launch_bounds__(256) void diag_kernel(const _Float16* __restrict__ xh,
                                                   float* __restrict__ D) {
    const int idx = blockIdx.x * 256 + threadIdx.x;   // 0 .. 7*4096-1
    const int m = idx >> 12;
    const int i = idx & (BSZ - 1);
    const _Float16* a = xh + (size_t)c_asel[m] * BSZ * KDIM + (size_t)i * KDIM;
    const _Float16* b = xh + (size_t)c_bsel[m] * BSZ * KDIM + (size_t)i * KDIM;
    float s = 0.0f;
    #pragma unroll
    for (int kc = 0; kc < KDIM / 8; ++kc) {
        const v8h av = *(const v8h*)(a + kc * 8);
        const v8h bv = *(const v8h*)(b + kc * 8);
        #pragma unroll
        for (int j = 0; j < 8; ++j) s += (float)av[j] * (float)bv[j];
    }
    D[idx] = fast_exp_sim(s);
}

// ---------------------------------------------------------------------------
// 4) Final scalar reduction: fold chunk partials, apply log terms, mean.
// ---------------------------------------------------------------------------
__global__ __launch_bounds__(256) void loss_final(const float* __restrict__ P,
                                                  const float* __restrict__ D,
                                                  float* __restrict__ out) {
    const int t = threadIdx.x;
    float local = 0.0f;
    for (int i = t; i < BSZ; i += 256) {
        float s[NMAT];
        #pragma unroll
        for (int m = 0; m < NMAT; ++m) {
            const float* p = P + (size_t)m * NCHUNK * BSZ + i;
            s[m] = p[0] + p[BSZ] + p[2 * BSZ] + p[3 * BSZ];
        }
        const float dxx = D[0 * BSZ + i], dyy = D[1 * BSZ + i], dxy = D[2 * BSZ + i];
        const float dax = D[3 * BSZ + i], day = D[4 * BSZ + i];
        const float dzx = D[5 * BSZ + i], dzy = D[6 * BSZ + i];

        const float denom = (s[2] - dxy) + (s[0] - dxx) + (s[1] - dyy);
        float li = -2.0f * logf(dxy / denom);
        li += -logf(dax / (s[3] - dax));
        li += -logf(day / (s[4] - day));
        li += -logf(dzx / (s[5] - dzx));
        li += -logf(dzy / (s[6] - dzy));
        local += li;
    }
    #pragma unroll
    for (int m = 16; m >= 1; m >>= 1) local += __shfl_xor(local, m, 32);
    __shared__ float red[8];
    const int wid = t >> 5, lane = t & 31;
    if (lane == 0) red[wid] = local;
    __syncthreads();
    if (t == 0) {
        float tot = 0.0f;
        #pragma unroll
        for (int w = 0; w < 8; ++w) tot += red[w];
        out[0] = tot / (float)BSZ;
    }
}

// ---------------------------------------------------------------------------
// Launch
// ---------------------------------------------------------------------------
extern "C" void kernel_launch(void* const* d_in, const int* in_sizes, int n_in,
                              void* d_out, int out_size, void* d_ws, size_t ws_size,
                              hipStream_t stream) {
    (void)in_sizes; (void)n_in; (void)out_size; (void)ws_size;
    const float* x = (const float*)d_in[0];
    float* out = (float*)d_out;

    char* ws = (char*)d_ws;
    _Float16* xh = (_Float16*)ws;                              // 12288*128*2 = 3,145,728 B
    float* P = (float*)(ws + (size_t)3 * BSZ * KDIM * 2);      // 7*4*4096*4  =   458,752 B
    float* D = P + (size_t)NMAT * NCHUNK * BSZ;                // 7*4096*4    =   114,688 B
                                                               // total ~3.55 MB of ws

    normalize_f16<<<3 * BSZ, 128, 0, stream>>>(x, xh);
    gemm_expsum<<<dim3(32 * NCHUNK, NMAT), 256, 0, stream>>>(xh, P);
    diag_kernel<<<NMAT * BSZ / 256, 256, 0, stream>>>(xh, D);
    loss_final<<<1, 256, 0, stream>>>(P, D, out);
}
